// BiMambaBlock_64347200029312
// MI455X (gfx1250) — matrix operand order repaired and checked by reading, not compiled, not run on hardware
//
#include <hip/hip_runtime.h>
#include <hip/hip_bf16.h>

// ---------------------------------------------------------------------------
// Types for CDNA5 WMMA
// ---------------------------------------------------------------------------
typedef __attribute__((ext_vector_type(16))) __bf16        v16bf;
typedef __attribute__((ext_vector_type(8)))  float         v8f;
typedef __attribute__((ext_vector_type(4)))  unsigned int  u32x4;

struct FragPair { u32x4 lo, hi; };   // 32 bytes == one v16bf fragment

// ---------------------------------------------------------------------------
// Problem constants
// ---------------------------------------------------------------------------
#define D_MODEL  1024
#define D_STATE  16
#define D_CONV   4
#define D_INNER  2048
#define DT_RANK  64
#define BATCH    2
#define SEQ      1024
#define ROWS     (BATCH * SEQ)      // 2048

#define BM 128
#define BN 128
#define BK 32
#define LS 40                       // LDS row stride in bf16 elems (80 B, 16B aligned)

// ---------------------------------------------------------------------------
// Generic bf16 WMMA GEMM:  C[M,N] (f32) = A[M,K] (bf16, row-major)
//                                       x W[N,K]^T (bf16, row-major)
// 128 threads = 4 waves; each wave computes a 64x64 quadrant via 4x4 grid of
// 16x16 WMMA accumulators (16 v_wmma per BK=32 step, 16 ds_load_b128).
// Double-buffered LDS + register staging: one barrier per K-step.
// act: 0 = none, 1 = softplus.  bias may be nullptr.
// ---------------------------------------------------------------------------
__launch_bounds__(128)
__global__ void k_gemm_bf16(const __bf16* __restrict__ A,
                            const __bf16* __restrict__ W,
                            float* __restrict__ C,
                            int M, int N, int K,
                            const float* __restrict__ bias, int act)
{
    __shared__ __bf16 As[2][BM * LS];   // 2 x 10240 B
    __shared__ __bf16 Bs[2][BN * LS];   // 2 x 10240 B

    const int tid  = threadIdx.x;        // 0..127
    const int m0   = blockIdx.y * BM;
    const int n0   = blockIdx.x * BN;
    const int wave = tid >> 5;           // 4 waves
    const int lane = tid & 31;
    const int wm   = (wave & 1) * 64;    // wave m-offset in tile
    const int wn   = (wave >> 1) * 64;   // wave n-offset in tile
    const int mrow = lane & 15;
    const int kbA  = (lane < 16) ? 0 : 8;    // A frag K base (ISA 16-bit A layout)
    const int khB  = (lane < 16) ? 0 : 8;    // B frag K base (same layout as A)

    const int nk = K / BK;
    const int nB = n0 + tid;             // B row this thread stages

    v8f acc[4][4];
#pragma unroll
    for (int i = 0; i < 4; ++i)
#pragma unroll
        for (int j = 0; j < 4; ++j)
#pragma unroll
            for (int e = 0; e < 8; ++e) acc[i][j][e] = 0.0f;

    u32x4 ra[4], rb[4];
    auto gload = [&](int k0) {
        const __bf16* sa = A + (size_t)(m0 + tid) * K + k0;
#pragma unroll
        for (int j = 0; j < 4; ++j) ra[j] = ((const u32x4*)sa)[j];
#pragma unroll
        for (int j = 0; j < 4; ++j) { u32x4 z = {0u,0u,0u,0u}; rb[j] = z; }
        if (nB < N) {
            const __bf16* sb = W + (size_t)nB * K + k0;
#pragma unroll
            for (int j = 0; j < 4; ++j) rb[j] = ((const u32x4*)sb)[j];
        }
    };
    auto lstore = [&](int buf) {
#pragma unroll
        for (int j = 0; j < 4; ++j)
            *((u32x4*)&As[buf][tid * LS + j * 8]) = ra[j];
#pragma unroll
        for (int j = 0; j < 4; ++j)
            *((u32x4*)&Bs[buf][tid * LS + j * 8]) = rb[j];
    };

    gload(0);
    lstore(0);

    for (int kt = 0; kt < nk; ++kt) {
        __syncthreads();                 // buffer `cur` ready; prior reads done
        const int cur  = kt & 1;
        const bool more = (kt + 1) < nk;
        if (more) gload((kt + 1) * BK);  // overlap next tile's global loads
        if (kt + 2 < nk) {               // L2 prefetch of tile k+2
            __builtin_prefetch(A + (size_t)(m0 + tid) * K + (kt + 2) * BK, 0, 1);
            if (nB < N)
                __builtin_prefetch(W + (size_t)nB * K + (kt + 2) * BK, 0, 1);
        }

        FragPair afr[4], bfr[4];
#pragma unroll
        for (int mt = 0; mt < 4; ++mt) {
            const __bf16* p = &As[cur][(wm + mt * 16 + mrow) * LS + kbA];
            afr[mt].lo = ((const u32x4*)p)[0];          // K = kbA .. kbA+7
            afr[mt].hi = *((const u32x4*)(p + 16));     // K = kbA+16 .. kbA+23
        }
#pragma unroll
        for (int nt = 0; nt < 4; ++nt) {
            const __bf16* p = &Bs[cur][(wn + nt * 16 + mrow) * LS + khB];
            bfr[nt].lo = ((const u32x4*)p)[0];          // K = khB .. khB+7
            bfr[nt].hi = *((const u32x4*)(p + 16));     // K = khB+16 .. khB+23
        }
#pragma unroll
        for (int mt = 0; mt < 4; ++mt) {
            v16bf av = __builtin_bit_cast(v16bf, afr[mt]);
#pragma unroll
            for (int nt = 0; nt < 4; ++nt) {
                v16bf bv = __builtin_bit_cast(v16bf, bfr[nt]);
                acc[mt][nt] = __builtin_amdgcn_wmma_f32_16x16x32_bf16(
                    false, av, false, bv, (short)0, acc[mt][nt], false, false);
            }
        }
        if (more) lstore((kt + 1) & 1);  // fill other buffer after reads issued
    }

    // ---- epilogue: ISA C/D layout (VGPR r: M = r + (lane<16?0:8), N = lane&15) ----
    const int rAdd = (lane < 16) ? 0 : 8;
    const int ncol = lane & 15;
#pragma unroll
    for (int mt = 0; mt < 4; ++mt) {
#pragma unroll
        for (int nt = 0; nt < 4; ++nt) {
            const int n = n0 + wn + nt * 16 + ncol;
            if (n >= N) continue;
            const float bv = bias ? bias[n] : 0.0f;
#pragma unroll
            for (int r = 0; r < 8; ++r) {
                const int m = m0 + wm + mt * 16 + r + rAdd;
                float v = acc[mt][nt][r] + bv;
                if (act == 1) v = (v > 20.0f) ? v : log1pf(__expf(v));
                C[(size_t)m * N + n] = v;
            }
        }
    }
}

// ---------------------------------------------------------------------------
// LayerNorm: x[2048,1024] -> bf16 normF (forward order) and normR
// (sequence reversed within each batch).
// ---------------------------------------------------------------------------
__launch_bounds__(256)
__global__ void k_layernorm(const float* __restrict__ x,
                            const float* __restrict__ w,
                            const float* __restrict__ b,
                            __bf16* __restrict__ nF, __bf16* __restrict__ nR)
{
    const int row = blockIdx.x;          // 0..2047
    const int t   = threadIdx.x;         // 256 threads
    const float* xr = x + (size_t)row * D_MODEL;

    float v[4], s = 0.f, ss = 0.f;
#pragma unroll
    for (int i = 0; i < 4; ++i) {
        float u = xr[t + 256 * i];
        v[i] = u; s += u; ss += u * u;
    }
    __shared__ float rs[256], rss[256];
    rs[t] = s; rss[t] = ss;
    __syncthreads();
    for (int off = 128; off > 0; off >>= 1) {
        if (t < off) { rs[t] += rs[t + off]; rss[t] += rss[t + off]; }
        __syncthreads();
    }
    const float mean = rs[0] * (1.0f / D_MODEL);
    const float var  = rss[0] * (1.0f / D_MODEL) - mean * mean;
    const float inv  = rsqrtf(var + 1e-5f);

    const int bb = row >> 10, l = row & 1023;
    const size_t rrow = (size_t)(bb * SEQ + (SEQ - 1 - l));
#pragma unroll
    for (int i = 0; i < 4; ++i) {
        const int c = t + 256 * i;
        const float o = (v[i] - mean) * inv * w[c] + b[c];
        nF[(size_t)row * D_MODEL + c] = (__bf16)o;
        nR[rrow * D_MODEL + c]        = (__bf16)o;
    }
}

// ---------------------------------------------------------------------------
// Depthwise causal conv (width 4) + bias + SiLU. xr: [2048, 2048] row=b*L+l.
// ---------------------------------------------------------------------------
__launch_bounds__(256)
__global__ void k_conv_silu(const float* __restrict__ xr,
                            const float* __restrict__ cw,
                            const float* __restrict__ cb,
                            float* __restrict__ xsO, __bf16* __restrict__ xsB)
{
    const int idx = blockIdx.x * 256 + threadIdx.x;
    if (idx >= ROWS * D_INNER) return;
    const int c = idx & (D_INNER - 1);
    const int m = idx >> 11;
    const int l = m & (SEQ - 1);
    const int base = m - l;              // b*SEQ
    float acc = cb[c];
#pragma unroll
    for (int j = 0; j < D_CONV; ++j) {
        const int ll = l - (D_CONV - 1) + j;
        if (ll >= 0) acc += xr[((size_t)(base + ll) << 11) + c] * cw[c * D_CONV + j];
    }
    const float sv = acc / (1.0f + __expf(-acc));
    xsO[idx] = sv;
    xsB[idx] = (__bf16)sv;
}

// ---------------------------------------------------------------------------
// Selective scan. One thread per (b, d, s). 16-lane shuffle reduction over s.
// dt:[2048,2048], xdb:[2048,96] (B at col 64, C at col 80), xs:[2048,2048].
// ---------------------------------------------------------------------------
__launch_bounds__(256)
__global__ void k_scan(const float* __restrict__ dt,
                       const float* __restrict__ xdb,
                       const float* __restrict__ xs,
                       const float* __restrict__ A_log,
                       float* __restrict__ y)
{
    const int t = threadIdx.x;
    const int s = t & 15;
    const int d = blockIdx.x * 16 + (t >> 4);
    const int b = blockIdx.y;
    const float a = -__expf(A_log[d * D_STATE + s]);
    float h = 0.0f;
    const size_t rowBase = (size_t)b * SEQ;
    for (int l = 0; l < SEQ; ++l) {
        const size_t r = rowBase + l;
        const float dtv = dt[(r << 11) + d];
        const float Bv  = xdb[r * 96 + DT_RANK + s];
        const float Cv  = xdb[r * 96 + DT_RANK + D_STATE + s];
        const float xv  = xs[(r << 11) + d];
        h = h * __expf(dtv * a) + dtv * Bv * xv;
        float p = h * Cv;
        p += __shfl_xor(p, 8, 16);
        p += __shfl_xor(p, 4, 16);
        p += __shfl_xor(p, 2, 16);
        p += __shfl_xor(p, 1, 16);
        if (s == 0) y[(r << 11) + d] = p;
    }
}

// ---------------------------------------------------------------------------
// Gate: out_bf16 = (y + xs*Dskip[c]) * silu(z)
// ---------------------------------------------------------------------------
__launch_bounds__(256)
__global__ void k_gate(const float* __restrict__ y, const float* __restrict__ xs,
                       const float* __restrict__ z, const float* __restrict__ Dskip,
                       __bf16* __restrict__ yg)
{
    const int idx = blockIdx.x * 256 + threadIdx.x;
    if (idx >= ROWS * D_INNER) return;
    const int c = idx & (D_INNER - 1);
    const float zv = z[idx];
    const float g  = zv / (1.0f + __expf(-zv));
    yg[idx] = (__bf16)((y[idx] + xs[idx] * Dskip[c]) * g);
}

// ---------------------------------------------------------------------------
// Strided f32 -> bf16 pack (optional per-batch row reversal), for
// combined-concat and the dt_lowrank slice of xdb.
// ---------------------------------------------------------------------------
__launch_bounds__(256)
__global__ void k_pack(const float* __restrict__ src, int srcLd, int srcColOff,
                       __bf16* __restrict__ dst, int dstLd, int dstColOff,
                       int cols, int rev, int total)
{
    const int idx = blockIdx.x * 256 + threadIdx.x;
    if (idx >= total) return;
    const int r = idx / cols;
    const int c = idx - r * cols;
    const int sr = rev ? ((r & ~(SEQ - 1)) + (SEQ - 1 - (r & (SEQ - 1)))) : r;
    dst[(size_t)r * dstLd + dstColOff + c] =
        (__bf16)src[(size_t)sr * srcLd + srcColOff + c];
}

// ---------------------------------------------------------------------------
// Plain f32 -> bf16 convert (weights)
// ---------------------------------------------------------------------------
__launch_bounds__(256)
__global__ void k_f2bf(const float* __restrict__ s, __bf16* __restrict__ d, int n)
{
    const int i = blockIdx.x * 256 + threadIdx.x;
    if (i < n) d[i] = (__bf16)s[i];
}

// ---------------------------------------------------------------------------
// Residual add: out += x
// ---------------------------------------------------------------------------
__launch_bounds__(256)
__global__ void k_residual(float* __restrict__ out, const float* __restrict__ x, int n)
{
    const int i = blockIdx.x * 256 + threadIdx.x;
    if (i < n) out[i] += x[i];
}

// ---------------------------------------------------------------------------
// Host launch
// ---------------------------------------------------------------------------
extern "C" void kernel_launch(void* const* d_in, const int* in_sizes, int n_in,
                              void* d_out, int out_size, void* d_ws, size_t ws_size,
                              hipStream_t stream)
{
    (void)in_sizes; (void)n_in; (void)ws_size;
    const float* x     = (const float*)d_in[0];
    const float* ln_w  = (const float*)d_in[1];
    const float* ln_b  = (const float*)d_in[2];
    const float* projW = (const float*)d_in[21];
    const float* projb = (const float*)d_in[22];
    float* out = (float*)d_out;

    // ---- workspace bump allocator (256B aligned) ----
    char* ws = (char*)d_ws;
    size_t off = 0;
    auto alloc = [&](size_t bytes) -> char* {
        char* p = ws + off;
        off = (off + bytes + 255) & ~(size_t)255;
        return p;
    };
    __bf16* wIn   = (__bf16*)alloc((size_t)2 * D_INNER * D_MODEL * 2);
    __bf16* wXp   = (__bf16*)alloc((size_t)(DT_RANK + 2 * D_STATE) * D_INNER * 2);
    __bf16* wDt   = (__bf16*)alloc((size_t)D_INNER * DT_RANK * 2);
    __bf16* wOut  = (__bf16*)alloc((size_t)D_MODEL * D_INNER * 2);
    __bf16* wProj = (__bf16*)alloc((size_t)D_MODEL * 2 * D_MODEL * 2);
    __bf16* normF = (__bf16*)alloc((size_t)ROWS * D_MODEL * 2);
    __bf16* normR = (__bf16*)alloc((size_t)ROWS * D_MODEL * 2);
    float*  xsraw = (float*)alloc((size_t)ROWS * D_INNER * 4);
    float*  zbuf  = (float*)alloc((size_t)ROWS * D_INNER * 4);
    float*  xsbuf = (float*)alloc((size_t)ROWS * D_INNER * 4);
    __bf16* xsbf  = (__bf16*)alloc((size_t)ROWS * D_INNER * 2);
    float*  xdb   = (float*)alloc((size_t)ROWS * 96 * 4);
    __bf16* dtlr  = (__bf16*)alloc((size_t)ROWS * DT_RANK * 2);
    float*  dtbuf = (float*)alloc((size_t)ROWS * D_INNER * 4);
    float*  ybuf  = (float*)alloc((size_t)ROWS * D_INNER * 4);
    __bf16* ygate = (__bf16*)alloc((size_t)ROWS * D_INNER * 2);
    float*  bout  = (float*)alloc((size_t)ROWS * D_MODEL * 4);
    __bf16* comb  = (__bf16*)alloc((size_t)ROWS * 2 * D_MODEL * 2);

    auto nblk = [](long n) { return (unsigned)((n + 255) / 256); };
    auto gemm = [&](const __bf16* A, const __bf16* W, float* C,
                    int M, int N, int K, const float* bias, int act) {
        dim3 g((N + BN - 1) / BN, M / BM);
        k_gemm_bf16<<<g, 128, 0, stream>>>(A, W, C, M, N, K, bias, act);
    };
    auto f2bf = [&](const float* s, __bf16* d, long n) {
        k_f2bf<<<nblk(n), 256, 0, stream>>>(s, d, (int)n);
    };

    // ---- shared layernorm (fwd + reversed) ----
    k_layernorm<<<ROWS, 256, 0, stream>>>(x, ln_w, ln_b, normF, normR);

    // ---- two mamba branches, sequential, buffers reused ----
    for (int dir = 0; dir < 2; ++dir) {
        const int base = 3 + dir * 9;
        const float* inW    = (const float*)d_in[base + 0];
        const float* convW  = (const float*)d_in[base + 1];
        const float* convb  = (const float*)d_in[base + 2];
        const float* xprojW = (const float*)d_in[base + 3];
        const float* dtW    = (const float*)d_in[base + 4];
        const float* dtb    = (const float*)d_in[base + 5];
        const float* A_log  = (const float*)d_in[base + 6];
        const float* Dskip  = (const float*)d_in[base + 7];
        const float* outW   = (const float*)d_in[base + 8];

        f2bf(inW,    wIn,  (long)2 * D_INNER * D_MODEL);
        f2bf(xprojW, wXp,  (long)(DT_RANK + 2 * D_STATE) * D_INNER);
        f2bf(dtW,    wDt,  (long)D_INNER * DT_RANK);
        f2bf(outW,   wOut, (long)D_MODEL * D_INNER);

        const __bf16* normed = dir ? normR : normF;

        // in-projection: xs half (rows 0..2047 of inW) and z half (rows 2048..4095)
        gemm(normed, wIn,                             xsraw, ROWS, D_INNER, D_MODEL, nullptr, 0);
        gemm(normed, wIn + (size_t)D_INNER * D_MODEL, zbuf,  ROWS, D_INNER, D_MODEL, nullptr, 0);

        // depthwise causal conv + SiLU
        k_conv_silu<<<nblk((long)ROWS * D_INNER), 256, 0, stream>>>(
            xsraw, convW, convb, xsbuf, xsbf);

        // x-projection -> [dt_lowrank | B | C]
        gemm(xsbf, wXp, xdb, ROWS, DT_RANK + 2 * D_STATE, D_INNER, nullptr, 0);

        // dt = softplus(dt_lowrank @ dtW^T + dtb)
        k_pack<<<nblk((long)ROWS * DT_RANK), 256, 0, stream>>>(
            xdb, 96, 0, dtlr, DT_RANK, 0, DT_RANK, 0, ROWS * DT_RANK);
        gemm(dtlr, wDt, dtbuf, ROWS, D_INNER, DT_RANK, dtb, 1);

        // selective scan
        k_scan<<<dim3(D_INNER / 16, BATCH), 256, 0, stream>>>(
            dtbuf, xdb, xsbuf, A_log, ybuf);

        // gate with z and D-skip
        k_gate<<<nblk((long)ROWS * D_INNER), 256, 0, stream>>>(
            ybuf, xsbuf, zbuf, Dskip, ygate);

        // out-projection
        gemm(ygate, wOut, bout, ROWS, D_MODEL, D_INNER, nullptr, 0);

        // pack into combined (backward branch un-reversed)
        k_pack<<<nblk((long)ROWS * D_MODEL), 256, 0, stream>>>(
            bout, D_MODEL, 0, comb, 2 * D_MODEL, dir * D_MODEL,
            D_MODEL, dir, ROWS * D_MODEL);
    }

    // ---- final projection + bias + residual ----
    f2bf(projW, wProj, (long)D_MODEL * 2 * D_MODEL);
    gemm(comb, wProj, out, ROWS, D_MODEL, 2 * D_MODEL, projb, 0);
    k_residual<<<nblk((long)out_size), 256, 0, stream>>>(out, x, out_size);
}
